// splineconv_37160057045414
// MI455X (gfx1250) — matrix operand order repaired and checked by reading, not compiled, not run on hardware
//
#include <hip/hip_runtime.h>
#include <hip/hip_bf16.h>
#include <math.h>

typedef __attribute__((ext_vector_type(16))) _Float16 v16h;
typedef __attribute__((ext_vector_type(8)))  _Float16 v8h;
typedef __attribute__((ext_vector_type(8)))  float    v8f;

constexpr int kN   = 4096;
constexpr int kE   = 16384;
constexpr int kDim = 6;
constexpr int kS   = 64;
constexpr int kES  = kE * kS;
constexpr int kKmax = 117649;   // 7^6

// ---------------------------------------------------------------------------
// Spline basis helpers (degree-1 open B-spline, matches reference)
// ---------------------------------------------------------------------------
static __device__ __forceinline__ void spline_prep(const float* __restrict__ ea,
                                                   int e, int ks,
                                                   float* frac, int* lo) {
#pragma unroll
  for (int d = 0; d < kDim; ++d) {
    float u  = ea[e * kDim + d];
    float v  = u * (float)(ks - 1);
    float fl = floorf(v);
    frac[d]  = v - fl;
    lo[d]    = (int)fl;
  }
}

__global__ void deg_kernel(const int* __restrict__ dst, float* __restrict__ deg) {
  int e = blockIdx.x * blockDim.x + threadIdx.x;
  if (e < kE) atomicAdd(&deg[dst[e]], 1.0f);
}

__global__ void hist_kernel(const float* __restrict__ ea, int* __restrict__ counts, int ks) {
  int e = blockIdx.x * blockDim.x + threadIdx.x;
  if (e >= kE) return;
  float frac[kDim]; int lo[kDim];
  spline_prep(ea, e, ks, frac, lo);
  for (int s = 0; s < kS; ++s) {
    int idx = 0, stride = 1;
#pragma unroll
    for (int d = 0; d < kDim; ++d) {
      int bit = (s >> d) & 1;
      int pos = lo[d] + bit;
      pos = pos < 0 ? 0 : (pos > ks - 1 ? ks - 1 : pos);
      idx += pos * stride;
      stride *= ks;
    }
    atomicAdd(&counts[idx], 1);
  }
}

// Single-workgroup chunked Hillis-Steele scan: counts[K] -> exclusive offs/cursor.
__global__ void scan_kernel(const int* __restrict__ counts, int* __restrict__ offs,
                            int* __restrict__ cursor, int K) {
  __shared__ int sm[1024];
  __shared__ int carry;
  if (threadIdx.x == 0) carry = 0;
  __syncthreads();
  for (int base = 0; base < K; base += 1024) {
    int i = base + (int)threadIdx.x;
    int v = (i < K) ? counts[i] : 0;
    sm[threadIdx.x] = v;
    __syncthreads();
    for (int off = 1; off < 1024; off <<= 1) {
      int t = (threadIdx.x >= (unsigned)off) ? sm[threadIdx.x - off] : 0;
      __syncthreads();
      sm[threadIdx.x] += t;
      __syncthreads();
    }
    int cbase = carry;
    int excl  = sm[threadIdx.x] - v;
    if (i < K) { offs[i] = cbase + excl; cursor[i] = cbase + excl; }
    __syncthreads();
    if (threadIdx.x == 0) carry = cbase + sm[1023];
    __syncthreads();
  }
}

__global__ void scatter_kernel(const float* __restrict__ ea, int* __restrict__ cursor,
                               int* __restrict__ pair_e, float* __restrict__ pair_b, int ks) {
  int e = blockIdx.x * blockDim.x + threadIdx.x;
  if (e >= kE) return;
  float frac[kDim]; int lo[kDim];
  spline_prep(ea, e, ks, frac, lo);
  for (int s = 0; s < kS; ++s) {
    int idx = 0, stride = 1;
    float basis = 1.0f;
#pragma unroll
    for (int d = 0; d < kDim; ++d) {
      int bit = (s >> d) & 1;
      int pos = lo[d] + bit;
      pos = pos < 0 ? 0 : (pos > ks - 1 ? ks - 1 : pos);
      idx += pos * stride;
      stride *= ks;
      basis *= bit ? frac[d] : (1.0f - frac[d]);
    }
    int p = atomicAdd(&cursor[idx], 1);
    pair_e[p] = e;
    pair_b[p] = basis;
  }
}

// ---------------------------------------------------------------------------
// Segmented GEMM: one workgroup per kernel index k.  rows = (edge,corner)
// pairs of the segment, cols = cout, reduction over cin via
// v_wmma_f32_16x16x32_f16 (f32 accumulate).  W[k] is streamed from HBM exactly
// once per layer; gathered node features come from the L2-resident node table.
// Results scatter straight into the per-node accumulator (sum over incoming
// edges), eliminating the intermediate per-edge message buffer.
// ---------------------------------------------------------------------------
template<int CIN, int CP, int CR, int NW>   // CP = cout padded, CR = real cout, NW = waves
__global__ __launch_bounds__(32 * NW)
void seg_gemm(const float* __restrict__ W, const float* __restrict__ h,
              const int* __restrict__ src, const int* __restrict__ dst,
              const int* __restrict__ pair_e, const float* __restrict__ pair_b,
              const int* __restrict__ offs, const int* __restrict__ counts,
              float* __restrict__ acc) {
  __shared__ __attribute__((aligned(32))) _Float16 ldsW[CP][CIN];  // B as [col][k]
  __shared__ __attribute__((aligned(32))) _Float16 ldsA[16][CIN];
  __shared__ int   tile_dst[16];
  __shared__ int   tile_src[16];   // pre-scaled by CIN
  __shared__ float tile_bs[16];

  constexpr int NT = 32 * NW;
  const int k = blockIdx.x;
  const int m = counts[k];
  if (m == 0) return;
  const int start = offs[k];
  const float* Wk = W + (size_t)k * CIN * CR;

  // Load W[k] (f32 row-major [cin][cout]) -> f16 LDS transposed [col][k]
  for (int i = threadIdx.x; i < CIN * CP; i += NT) {
    int kk = i / CP, col = i - kk * CP;
    float v = (col < CR) ? Wk[kk * CR + col] : 0.0f;
    ldsW[col][kk] = (_Float16)v;
  }

  const int wave    = threadIdx.x >> 5;
  const int lane    = threadIdx.x & 31;
  const int half_hi = lane >> 4;     // 0: lanes 0-15, 1: lanes 16-31
  const int ln16    = lane & 15;

  for (int r0 = 0; r0 < m; r0 += 16) {
    __syncthreads();
    // Stage tile metadata (one global read per row, shared via LDS)
    if (threadIdx.x < 16) {
      int row = threadIdx.x;
      int d_ = -1, s_ = 0;
      float bs = 0.0f;
      if (r0 + row < m) {
        int p = start + r0 + row;
        int e = pair_e[p];
        bs = pair_b[p];
        s_ = src[e] * CIN;
        d_ = dst[e];
      }
      tile_dst[row] = d_;
      tile_src[row] = s_;
      tile_bs[row]  = bs;
    }
    __syncthreads();
    // Build A tile: a[row][c] = basis * x[src[e]][c]  (f16; basis=0 pads rows)
    for (int i = threadIdx.x; i < 16 * CIN; i += NT) {
      int row = i / CIN, c = i - row * CIN;
      ldsA[row][c] = (_Float16)(tile_bs[row] * h[tile_src[row] + c]);
    }
    __syncthreads();

    // Per-lane destination rows (from LDS, hoisted out of the column loop)
    int myDst[8];
#pragma unroll
    for (int r = 0; r < 8; ++r) myDst[r] = tile_dst[half_hi * 8 + r];
    const bool full = (r0 + 16 <= m);

    for (int ct = wave; ct < CP / 16; ct += NW) {
      v8f d = {};
#pragma unroll
      for (int k0 = 0; k0 < CIN; k0 += 32) {
        // A lane layout: 16 contiguous K starting at k0 + hi*16
        const int kA = k0 + half_hi * 16;
        v8h alo = *(const v8h*)&ldsA[ln16][kA];
        v8h ahi = *(const v8h*)&ldsA[ln16][kA + 8];
        v16h af = __builtin_shufflevector(alo, ahi,
                    0,1,2,3,4,5,6,7,8,9,10,11,12,13,14,15);
        // B lane layout: 16 contiguous K starting at k0 + hi*16, column = lane%16
        v16h bf = *(const v16h*)&ldsW[ct * 16 + ln16][k0 + half_hi * 16];
        d = __builtin_amdgcn_wmma_f32_16x16x32_f16(
                false, af, false, bf, (short)0, d, false, false);
      }
      // D: reg r -> row r + 8*half_hi, col = lane%16; scatter-add to nodes.
      const int n = ct * 16 + ln16;
      if ((CR % 16 == 0) || n < CR) {
        if (full) {
#pragma unroll
          for (int r = 0; r < 8; ++r)
            atomicAdd(&acc[(size_t)myDst[r] * CR + n], d[r]);
        } else {
#pragma unroll
          for (int r = 0; r < 8; ++r)
            if (myDst[r] >= 0)
              atomicAdd(&acc[(size_t)myDst[r] * CR + n], d[r]);
        }
      }
    }
  }
}

__global__ void finalize_kernel(const float* __restrict__ acc, const float* __restrict__ deg,
                                const float* __restrict__ h, const float* __restrict__ Wr,
                                const float* __restrict__ b, float* __restrict__ out,
                                int cin, int cout) {
  int i = blockIdx.x * blockDim.x + threadIdx.x;
  if (i >= kN * cout) return;
  int n = i / cout, c = i - n * cout;
  float s = acc[i] / fmaxf(deg[n], 1.0f);
  float dot = 0.0f;
  for (int kk = 0; kk < cin; ++kk) dot += h[(size_t)n * cin + kk] * Wr[kk * cout + c];
  out[i] = s + dot + b[c];
}

__global__ void elu_lsm_kernel(const float* __restrict__ in, float* __restrict__ out) {
  int n = blockIdx.x * blockDim.x + threadIdx.x;
  if (n >= kN) return;
  float v[13];
  float mx = -3.0e38f;
  for (int c = 0; c < 13; ++c) {
    float x = in[n * 13 + c];
    x = x > 0.0f ? x : (expf(x) - 1.0f);
    v[c] = x;
    mx = fmaxf(mx, x);
  }
  float s = 0.0f;
  for (int c = 0; c < 13; ++c) s += expf(v[c] - mx);
  float lse = logf(s);
  for (int c = 0; c < 13; ++c) out[n * 13 + c] = v[c] - mx - lse;
}

// ---------------------------------------------------------------------------
// Host side
// ---------------------------------------------------------------------------
template<int CIN, int CP, int CR, int NW>
static void run_conv(const float* h_in, const float* W, const float* Wr, const float* b,
                     float* h_out, int K,
                     const int* src, const int* dst,
                     const int* pair_e, const float* pair_b,
                     const int* offs, const int* counts,
                     float* acc, const float* deg, hipStream_t stream) {
  hipMemsetAsync(acc, 0, (size_t)kN * CR * sizeof(float), stream);
  seg_gemm<CIN, CP, CR, NW><<<dim3(K), dim3(32 * NW), 0, stream>>>(
      W, h_in, src, dst, pair_e, pair_b, offs, counts, acc);
  finalize_kernel<<<(kN * CR + 255) / 256, 256, 0, stream>>>(
      acc, deg, h_in, Wr, b, h_out, CIN, CR);
}

extern "C" void kernel_launch(void* const* d_in, const int* in_sizes, int n_in,
                              void* d_out, int out_size, void* d_ws, size_t ws_size,
                              hipStream_t stream) {
  (void)in_sizes; (void)n_in; (void)out_size; (void)ws_size;
  const float* x    = (const float*)d_in[0];
  const int*   eidx = (const int*)d_in[1];
  const float* ea   = (const float*)d_in[2];
  const float* W1  = (const float*)d_in[3];
  const float* Wr1 = (const float*)d_in[4];
  const float* b1  = (const float*)d_in[5];
  const float* W3  = (const float*)d_in[6];
  const float* Wr3 = (const float*)d_in[7];
  const float* b3  = (const float*)d_in[8];
  const float* W6  = (const float*)d_in[9];
  const float* Wr6 = (const float*)d_in[10];
  const float* b6  = (const float*)d_in[11];
  const float* W7  = (const float*)d_in[12];
  const float* Wr7 = (const float*)d_in[13];
  const float* b7  = (const float*)d_in[14];
  const int* src = eidx;
  const int* dst = eidx + kE;
  float* out = (float*)d_out;

  // Workspace carve-out (~16 MB total)
  char* ws = (char*)d_ws;
  auto carve = [&](size_t bytes) {
    char* p = ws;
    ws += (bytes + 255) & ~(size_t)255;
    return p;
  };
  float* hA     = (float*)carve((size_t)kN * 128 * 4);
  float* hB     = (float*)carve((size_t)kN * 128 * 4);
  float* acc    = (float*)carve((size_t)kN * 128 * 4);
  float* deg    = (float*)carve((size_t)kN * 4);
  int*   counts = (int*)carve((size_t)kKmax * 4);
  int*   offs   = (int*)carve((size_t)kKmax * 4);
  int*   cursor = (int*)carve((size_t)kKmax * 4);
  int*   pair_e = (int*)carve((size_t)kES * 4);
  float* pair_b = (float*)carve((size_t)kES * 4);

  // In-degree (shared by all layers)
  hipMemsetAsync(deg, 0, kN * sizeof(float), stream);
  deg_kernel<<<(kE + 255) / 256, 256, 0, stream>>>(dst, deg);

  auto build_pairs = [&](int ks, int K) {
    hipMemsetAsync(counts, 0, (size_t)K * sizeof(int), stream);
    hist_kernel<<<(kE + 255) / 256, 256, 0, stream>>>(ea, counts, ks);
    scan_kernel<<<1, 1024, 0, stream>>>(counts, offs, cursor, K);
    scatter_kernel<<<(kE + 255) / 256, 256, 0, stream>>>(ea, cursor, pair_e, pair_b, ks);
  };

  // Layers 1 & 3 share ks=3 pair ordering (729 kernel weights)
  build_pairs(3, 729);
  run_conv<32, 128, 128, 4>(x,  W1, Wr1, b1, hA, 729,
      src, dst, pair_e, pair_b, offs, counts, acc, deg, stream);
  run_conv<128, 128, 128, 4>(hA, W3, Wr3, b3, hB, 729,
      src, dst, pair_e, pair_b, offs, counts, acc, deg, stream);

  build_pairs(5, 15625);
  run_conv<128, 64, 64, 4>(hB, W6, Wr6, b6, hA, 15625,
      src, dst, pair_e, pair_b, offs, counts, acc, deg, stream);

  build_pairs(7, 117649);
  run_conv<64, 16, 13, 1>(hA, W7, Wr7, b7, hB, 117649,
      src, dst, pair_e, pair_b, offs, counts, acc, deg, stream);

  elu_lsm_kernel<<<(kN + 255) / 256, 256, 0, stream>>>(hB, out);
}
